// DeepseekV4Indexer_18425409700452
// MI455X (gfx1250) — compile-verified, matrix operand order checked by reading, not run on hardware
//
#include <hip/hip_runtime.h>
#include <hip/hip_bf16.h>
#include <math.h>

// ---------------- problem constants ----------------
#define B_   2
#define S_   4096
#define HID_ 2048
#define QLR_ 1536
#define H_   16
#define HD_  128
#define RD_  64
#define RATIO_ 4
#define TOPK_ 512
#define P_   (S_ / RATIO_)          // 1024
#define EPS_ 1e-6f
#define THETA_ 10000.0f
#define SCALE_ 0.08838834764831845f // 128^-0.5

typedef __attribute__((ext_vector_type(16))) __bf16 v16bf;
typedef __attribute__((ext_vector_type(8)))  __bf16 bf16x8;
typedef __attribute__((ext_vector_type(8)))  float  v8f;
typedef __attribute__((ext_vector_type(4)))  unsigned int u32x4;
typedef __attribute__((ext_vector_type(8)))  int i32x8;
typedef __attribute__((ext_vector_type(4)))  int i32x4;

#if defined(__has_builtin)
#if __has_builtin(__builtin_amdgcn_tensor_load_to_lds) && \
    __has_builtin(__builtin_amdgcn_s_wait_tensorcnt)
#define HAVE_TDM_ 1
#endif
#endif
#ifndef HAVE_TDM_
#define HAVE_TDM_ 0
#endif

// ---------------------------------------------------------------------------
// WMMA bf16 fragment loaders.
// A (16x32, row-major, 16-bit): lane<16 -> M=lane, elems 0..7 = K 0..7,
// elems 8..15 = K 16..23 ; lane>=16 -> M=lane-16, K 8..15 / 24..31.
// B fragments use the transposed weight (N x K row-major) with lane<->column,
// same K striping.
// ---------------------------------------------------------------------------
__device__ __forceinline__ v16bf frag_ld(const __bf16* base, int stride, int k0) {
    const int lane = threadIdx.x & 31;
    const int row  = lane & 15;
    const int koff = (lane < 16) ? 0 : 8;
    const __bf16* p = base + (long)row * stride + k0 + koff;
    bf16x8 lo = *(const bf16x8*)(p);
    bf16x8 hi = *(const bf16x8*)(p + 16);
    v16bf r;
#pragma unroll
    for (int i = 0; i < 8; ++i) { r[i] = lo[i]; r[i + 8] = hi[i]; }
    return r;
}

// A fragment from an explicit per-lane row pointer (for clamped rows)
__device__ __forceinline__ v16bf frag_ld_ptr(const __bf16* rowptr, int k0) {
    const int lane = threadIdx.x & 31;
    const int koff = (lane < 16) ? 0 : 8;
    const __bf16* p = rowptr + k0 + koff;
    bf16x8 lo = *(const bf16x8*)(p);
    bf16x8 hi = *(const bf16x8*)(p + 16);
    v16bf r;
#pragma unroll
    for (int i = 0; i < 8; ++i) { r[i] = lo[i]; r[i + 8] = hi[i]; }
    return r;
}

__device__ __forceinline__ v8f wmma_bf16(v16bf a, v16bf b, v8f c) {
    return __builtin_amdgcn_wmma_f32_16x16x32_bf16(false, a, false, b,
                                                   (short)0, c, false, false);
}

// C/D layout: vgpr v, lane l -> M = v + (l>=16 ? 8 : 0), N = l & 15.
__device__ __forceinline__ int cd_row(int v) {
    return v + (((threadIdx.x & 31) >= 16) ? 8 : 0);
}

// ---------------------------------------------------------------------------
// Kernel 1a: fp32 -> bf16 (grid stride)
// ---------------------------------------------------------------------------
__global__ void cvt_bf16_kernel(const float* __restrict__ src,
                                __bf16* __restrict__ dst, long n) {
    for (long i = (long)blockIdx.x * blockDim.x + threadIdx.x; i < n;
         i += (long)gridDim.x * blockDim.x)
        dst[i] = (__bf16)src[i];
}

// Kernel 1b: fp32 (rows x cols) -> bf16 transposed (cols x rows)
__global__ void cvt_t_bf16_kernel(const float* __restrict__ src,
                                  __bf16* __restrict__ dst, int rows, int cols) {
    long n = (long)rows * cols;
    for (long i = (long)blockIdx.x * blockDim.x + threadIdx.x; i < n;
         i += (long)gridDim.x * blockDim.x) {
        int r = (int)(i / cols), c = (int)(i % cols);
        dst[(long)c * rows + r] = (__bf16)src[i];
    }
}

// ---------------------------------------------------------------------------
// Kernel 2: kv/gate GEMM + pooling for TWO consecutive p-blocks per workgroup.
// A tile = 16 hidden rows s in [4*p0-4, 4*p0+12): p0 uses local rows 0..7,
// p0+1 uses local rows 4..11 (rows 12..15 are pad). N=256, K=2048.
// 512 threads = 16 waves; waves 0-7 do kv, 8-15 do gate; each wave 2 N-tiles.
// Epilogue: overlap softmax pooling, RMSNorm, partial RoPE -> k_bf16.
// ---------------------------------------------------------------------------
__global__ void pool_k_kernel(const __bf16* __restrict__ hiddenB,
                              const __bf16* __restrict__ wkvT,   // 256 x 2048
                              const __bf16* __restrict__ wgateT, // 256 x 2048
                              const float*  __restrict__ ape,    // 4 x 256
                              const float*  __restrict__ kv_norm_w,
                              __bf16* __restrict__ k_bf16) {
    __shared__ float kvS[16 * 256];
    __shared__ float gateS[16 * 256];
    __shared__ float redS[256];
    __shared__ float knS[256];

    const int blk = blockIdx.x;
    const int b  = blk / (P_ / 2);
    const int p0 = (blk % (P_ / 2)) * 2;
    const int wid  = threadIdx.x >> 5;
    const int lane = threadIdx.x & 31;
    const int mat  = wid >> 3;          // 0 = kv, 1 = gate
    const int nt2  = wid & 7;           // pair of 16-col tiles

    // per-lane A row, clamped at both ends (pads are masked in the epilogue)
    long srow = (long)4 * p0 - 4 + (lane & 15);
    if (srow < 0) srow = 0;
    if (srow > S_ - 1) srow = S_ - 1;
    const __bf16* aRow = hiddenB + ((long)b * S_ + srow) * HID_;
    const __bf16* Bt = mat ? wgateT : wkvT;
    const int nb0 = 16 * (2 * nt2);
    const int nb1 = nb0 + 16;

    v8f acc0 = {}, acc1 = {};
#pragma unroll 4
    for (int kt = 0; kt < HID_ / 32; ++kt) {
        const int k0 = kt * 32;
        __builtin_prefetch(aRow + k0 + 256, 0, 1);
        v16bf a  = frag_ld_ptr(aRow, k0);
        v16bf b0 = frag_ld(Bt + (long)nb0 * HID_, HID_, k0);
        v16bf b1 = frag_ld(Bt + (long)nb1 * HID_, HID_, k0);
        acc0 = wmma_bf16(a, b0, acc0);
        acc1 = wmma_bf16(a, b1, acc1);
    }
    float* dstArr = mat ? gateS : kvS;
#pragma unroll
    for (int v = 0; v < 8; ++v) {
        int m = cd_row(v), n = lane & 15;
        dstArr[m * 256 + nb0 + n] = acc0[v];
        dstArr[m * 256 + nb1 + n] = acc1[v];
    }
    __syncthreads();

    // softmax pooling over 8 overlap candidates; thread t -> (pi = t>>7, d)
    const int t = threadIdx.x;
    const int pi = t >> 7;              // which of the two p-blocks
    const int d  = t & 127;
    const int p  = p0 + pi;
    float pooled = 0.f;
    if (t < 256) {
        const int rb = 4 * pi;          // local row base for this p
        float g[8], kv[8];
#pragma unroll
        for (int c = 0; c < 8; ++c) {
            const int lr = rb + c;
            if (c < 4) {
                if (p == 0) { g[c] = -INFINITY; kv[c] = 0.f; }
                else { g[c] = gateS[lr * 256 + d] + ape[c * 256 + d];
                       kv[c] = kvS[lr * 256 + d]; }
            } else {
                g[c] = gateS[lr * 256 + 128 + d] + ape[(c - 4) * 256 + 128 + d];
                kv[c] = kvS[lr * 256 + 128 + d];
            }
        }
        float mx = -INFINITY;
#pragma unroll
        for (int c = 0; c < 8; ++c) mx = fmaxf(mx, g[c]);
        float den = 0.f, num = 0.f;
#pragma unroll
        for (int c = 0; c < 8; ++c) {
            float e = __expf(g[c] - mx);
            den += e; num += e * kv[c];
        }
        pooled = num / den;
        redS[t] = pooled * pooled;
    }
    __syncthreads();
    for (int off = 64; off > 0; off >>= 1) {
        if (t < 256 && (t & 127) < off) redS[t] += redS[t + off];
        __syncthreads();
    }
    if (t < 256) {
        const float rms = rsqrtf(redS[pi * 128] * (1.0f / HD_) + EPS_);
        knS[t] = pooled * rms * kv_norm_w[d];
    }
    __syncthreads();

    if (t < 256) {
        __bf16* kout = k_bf16 + ((long)b * P_ + p) * HD_;
        if (d < 64) {                      // nope half passes through
            kout[d] = (__bf16)knS[pi * 128 + d];
        } else if (d < 96) {               // rope half: 32 pairs
            const int i = d - 64;
            const float pos = (float)(p * RATIO_);
            const float ang = pos * __powf(THETA_, -(float)(2 * i) / (float)RD_);
            const float c = __cosf(ang), s = __sinf(ang);
            const float a  = knS[pi * 128 + 64 + 2 * i];
            const float bb = knS[pi * 128 + 64 + 2 * i + 1];
            kout[64 + 2 * i]     = (__bf16)(a * c - bb * s);
            kout[64 + 2 * i + 1] = (__bf16)(a * s + bb * c);
        }
    }
}

// ---------------------------------------------------------------------------
// Kernel 3: q projection GEMM (8192 x 2048 x 1536) + partial RoPE epilogue.
// 256 threads = 8 waves in a 2x4 grid; block tile 32(M) x 64(N).
// ---------------------------------------------------------------------------
__global__ void q_proj_kernel(const __bf16* __restrict__ qresB, // 8192 x 1536
                              const __bf16* __restrict__ wqbT,  // 2048 x 1536
                              __bf16* __restrict__ qB) {        // 8192 x 2048
    __shared__ float tileS[32][64];
    const int bm = blockIdx.x & 255;   // 8192/32 = 256 M tiles
    const int bn = blockIdx.x >> 8;    // 2048/64 = 32 N tiles
    const int wid  = threadIdx.x >> 5;
    const int lane = threadIdx.x & 31;
    const int wm = wid >> 2, wn = wid & 3;

    const long row0 = (long)bm * 32 + wm * 16;
    const int  col0 = bn * 64 + wn * 16;
    const __bf16* aBase = qresB + row0 * QLR_;
    const __bf16* bBase = wqbT + (long)col0 * QLR_;

    v8f acc = {};
#pragma unroll 4
    for (int kt = 0; kt < QLR_ / 32; ++kt) {
        const int k0 = kt * 32;
        __builtin_prefetch(aBase + ((lane & 15) * QLR_) + k0 + 256, 0, 1);
        v16bf a = frag_ld(aBase, QLR_, k0);
        v16bf b = frag_ld(bBase, QLR_, k0);
        acc = wmma_bf16(a, b, acc);
    }
#pragma unroll
    for (int v = 0; v < 8; ++v)
        tileS[wm * 16 + cd_row(v)][wn * 16 + (lane & 15)] = acc[v];
    __syncthreads();

    // RoPE + store (32*64 = 2048 elems, 8 per thread)
#pragma unroll
    for (int e = 0; e < 8; ++e) {
        const int idx = threadIdx.x + e * 256;
        const int r = idx >> 6, j = idx & 63;
        const long grow = (long)bm * 32 + r;
        const int  gcol = bn * 64 + j;
        const int  d = gcol & (HD_ - 1);
        float val;
        if (d < RD_) {
            val = tileS[r][j];
        } else {
            const int i = (d - RD_) >> 1;
            const float pos = (float)(grow & (S_ - 1));
            const float ang = pos * __powf(THETA_, -(float)(2 * i) / (float)RD_);
            const float c = __cosf(ang), s = __sinf(ang);
            const float a = tileS[r][j & ~1], bb = tileS[r][j | 1];
            val = (d & 1) ? (a * s + bb * c) : (a * c - bb * s);
        }
        qB[grow * (long)(H_ * HD_) + gcol] = (__bf16)val;
    }
}

// ---------------------------------------------------------------------------
// Kernel 4: weights = hidden @ weights_proj (8192 x 16 x 2048); 1 wave/block.
// ---------------------------------------------------------------------------
__global__ void weights_proj_kernel(const __bf16* __restrict__ hiddenB,
                                    const __bf16* __restrict__ wprojT, // 16 x 2048
                                    float* __restrict__ weights) {     // 8192 x 16
    const int lane = threadIdx.x & 31;
    const long row0 = (long)blockIdx.x * 16;
    const __bf16* aBase = hiddenB + row0 * HID_;
    v8f acc = {};
#pragma unroll 4
    for (int kt = 0; kt < HID_ / 32; ++kt) {
        v16bf a = frag_ld(aBase, HID_, kt * 32);
        v16bf b = frag_ld(wprojT, HID_, kt * 32);
        acc = wmma_bf16(a, b, acc);
    }
#pragma unroll
    for (int v = 0; v < 8; ++v)
        weights[(row0 + cd_row(v)) * H_ + (lane & 15)] = acc[v];
}

// ---------------------------------------------------------------------------
// Kernel 5: masked score. The 128x128 bf16 k-tile shared by all 8 waves is
// staged into LDS once via the Tensor Data Mover (tensor_load_to_lds +
// s_wait_tensorcnt), then every wave frag-loads it with ds_load_b128.
// Per wave: a 16(s) x 16(p) tile; 4 K-frags of k reused across 16 heads
// (64 WMMAs/tile); relu + head weights from LDS; scale + causal mask.
// ---------------------------------------------------------------------------
__global__ void score_kernel(const __bf16* __restrict__ qB,     // 8192 x 2048
                             const __bf16* __restrict__ kB,     // B*P x 128
                             const float*  __restrict__ weights,// 8192 x 16
                             float* __restrict__ out_masked) {  // B*S x P
    __shared__ float wS[16 * 16];
    __shared__ __bf16 kTile[128 * HD_];      // 128 rows x 128 cols = 32 KB
    const int pb = blockIdx.x & 7;           // P/128
    const int sb = (blockIdx.x >> 3) & 255;  // S/16
    const int b  = blockIdx.x >> 11;
    const int wid  = threadIdx.x >> 5;
    const int lane = threadIdx.x & 31;
    const int s0 = sb * 16;
    const int p0 = pb * 128 + wid * 16;

    wS[threadIdx.x] = weights[((long)b * S_ + s0 + (threadIdx.x >> 4)) * H_ +
                              (threadIdx.x & 15)];

    const __bf16* kSrc = kB + ((long)b * P_ + pb * 128) * HD_;
#if HAVE_TDM_
    if (threadIdx.x < 32) {
        // Tensor DMA descriptor (ISA 8.3/8.4): 2-D tile 128x128, 2B elems.
        const unsigned long long gaddr = (unsigned long long)(const void*)kSrc;
        const unsigned ldsa = (unsigned)(uintptr_t)(void*)kTile;
        u32x4 g0 = {};
        g0[0] = 1u;                                    // count = 1
        g0[1] = ldsa;                                  // lds_addr
        g0[2] = (unsigned)gaddr;                       // global_addr[31:0]
        g0[3] = (unsigned)((gaddr >> 32) & 0x1FFFFFFu) // global_addr[56:32]
                | (2u << 30);                          // type = 2 (image)
        i32x8 g1 = {};
        g1[0] = (int)(1u << 16);                       // data_size = 2 bytes
        g1[1] = (int)(128u << 16);                     // tensor_dim0 = 128
        g1[2] = (int)(128u << 16);                     // tensor_dim1 = 128
        g1[3] = (int)(128u << 16);                     // tile_dim0 = 128
        g1[4] = 128;                                   // tile_dim1 = 128
        g1[5] = 128;                                   // tensor_dim0_stride
        g1[6] = 0;
        g1[7] = 0;
        i32x4 g2 = {}, g3 = {};
        i32x8 g4 = {};                                 // extra group (6-arg form)
        __builtin_amdgcn_tensor_load_to_lds(g0, g1, g2, g3, g4, 0);
        __builtin_amdgcn_s_wait_tensorcnt(0);
    }
#else
    // fallback: cooperative copy, 128 B per thread
#pragma unroll
    for (int e = 0; e < 8; ++e) {
        const int idx = threadIdx.x + e * 256;           // 16B chunks
        ((bf16x8*)kTile)[idx] = ((const bf16x8*)kSrc)[idx];
    }
#endif
    __syncthreads();

    // B fragments from LDS (wave's 16 k-rows)
    const __bf16* kRows = kTile + wid * 16 * HD_;
    v16bf bk[4];
#pragma unroll
    for (int kt = 0; kt < 4; ++kt) bk[kt] = frag_ld(kRows, HD_, kt * 32);

    const __bf16* qBase = qB + ((long)b * S_ + s0) * (long)(H_ * HD_);
    v8f sc = {};
    for (int h = 0; h < H_; ++h) {
        v8f acc = {};
#pragma unroll
        for (int kt = 0; kt < 4; ++kt) {
            v16bf a = frag_ld(qBase, H_ * HD_, h * HD_ + kt * 32);
            acc = wmma_bf16(a, bk[kt], acc);
        }
#pragma unroll
        for (int v = 0; v < 8; ++v) {
            const float w = wS[cd_row(v) * 16 + h];
            sc[v] += w * fmaxf(acc[v], 0.f);
        }
    }
#pragma unroll
    for (int v = 0; v < 8; ++v) {
        const int sg = s0 + cd_row(v);
        const int pg = p0 + (lane & 15);
        const float val = (sg >= pg * RATIO_ + RATIO_ - 1) ? sc[v] * SCALE_
                                                           : -INFINITY;
        out_masked[((long)b * S_ + sg) * P_ + pg] = val;
    }
}

// ---------------------------------------------------------------------------
// Kernel 6: per-row top-512 of 1024 via bitonic sort (desc val, asc idx).
// ---------------------------------------------------------------------------
__global__ void topk_kernel(const float* __restrict__ masked,
                            int* __restrict__ out_idx) {
    __shared__ float sv[P_];
    __shared__ int   si[P_];
    const long row = blockIdx.x;
    const int  t = threadIdx.x;
    const float* mrow = masked + row * P_;
    sv[t] = mrow[t];             si[t] = t;
    sv[t + 512] = mrow[t + 512]; si[t + 512] = t + 512;

    for (unsigned k = 2; k <= P_; k <<= 1) {
        for (unsigned j = k >> 1; j > 0; j >>= 1) {
            __syncthreads();
            const unsigned i = ((t & ~(j - 1)) << 1) | (t & (j - 1));
            const unsigned pp = i | j;
            const float va = sv[i], vb = sv[pp];
            const int   ia = si[i], ib = si[pp];
            const bool aFirst = (va > vb) || (va == vb && ia < ib);
            const bool desc = ((i & k) == 0);
            if (desc ? !aFirst : aFirst) {
                sv[i] = vb; sv[pp] = va; si[i] = ib; si[pp] = ia;
            }
        }
    }
    __syncthreads();
    out_idx[row * TOPK_ + t] = si[t];
}

// ---------------------------------------------------------------------------
// launcher
// ---------------------------------------------------------------------------
extern "C" void kernel_launch(void* const* d_in, const int* in_sizes, int n_in,
                              void* d_out, int out_size, void* d_ws, size_t ws_size,
                              hipStream_t stream) {
    (void)in_sizes; (void)n_in; (void)out_size; (void)ws_size;
    const float* hidden  = (const float*)d_in[0]; // B,S,HID
    const float* qres    = (const float*)d_in[1]; // B,S,QLR
    const float* wkv     = (const float*)d_in[2]; // HID, 256
    const float* wgate   = (const float*)d_in[3]; // HID, 256
    const float* ape     = (const float*)d_in[4]; // 4, 256
    const float* kvnorm  = (const float*)d_in[5]; // 128
    const float* wqb     = (const float*)d_in[6]; // QLR, 2048
    const float* wproj   = (const float*)d_in[7]; // HID, 16

    // workspace carve-up (256B aligned)
    char* w = (char*)d_ws;
    auto carve = [&](size_t bytes) {
        char* r = w; w += (bytes + 255) & ~(size_t)255; return r;
    };
    __bf16* hiddenB = (__bf16*)carve((size_t)B_ * S_ * HID_ * 2);
    __bf16* qresB   = (__bf16*)carve((size_t)B_ * S_ * QLR_ * 2);
    __bf16* wkvT    = (__bf16*)carve((size_t)256 * HID_ * 2);
    __bf16* wgateT  = (__bf16*)carve((size_t)256 * HID_ * 2);
    __bf16* wqbT    = (__bf16*)carve((size_t)(H_ * HD_) * QLR_ * 2);
    __bf16* wprojT  = (__bf16*)carve((size_t)H_ * HID_ * 2);
    __bf16* kB      = (__bf16*)carve((size_t)B_ * P_ * HD_ * 2);
    __bf16* qB      = (__bf16*)carve((size_t)B_ * S_ * H_ * HD_ * 2);
    float*  weights = (float*) carve((size_t)B_ * S_ * H_ * 4);

    int*   out_idx    = (int*)d_out;                               // B,S,TOPK
    float* out_masked = (float*)d_out + (size_t)B_ * S_ * TOPK_;   // B,S,P

    // 1) precision staging
    cvt_bf16_kernel<<<4096, 256, 0, stream>>>(hidden, hiddenB, (long)B_ * S_ * HID_);
    cvt_bf16_kernel<<<4096, 256, 0, stream>>>(qres, qresB, (long)B_ * S_ * QLR_);
    cvt_t_bf16_kernel<<<2048, 256, 0, stream>>>(wkv,   wkvT,   HID_, 256);
    cvt_t_bf16_kernel<<<2048, 256, 0, stream>>>(wgate, wgateT, HID_, 256);
    cvt_t_bf16_kernel<<<2048, 256, 0, stream>>>(wqb,   wqbT,   QLR_, H_ * HD_);
    cvt_t_bf16_kernel<<<256,  256, 0, stream>>>(wproj, wprojT, HID_, H_);

    // 2) kv/gate GEMM + overlap softmax pooling + RMSNorm + RoPE -> k
    pool_k_kernel<<<B_ * (P_ / 2), 512, 0, stream>>>(hiddenB, wkvT, wgateT, ape,
                                                     kvnorm, kB);
    // 3) q projection + RoPE
    q_proj_kernel<<<256 * 32, 256, 0, stream>>>(qresB, wqbT, qB);
    // 4) head-weight projection
    weights_proj_kernel<<<(B_ * S_) / 16, 32, 0, stream>>>(hiddenB, wprojT, weights);
    // 5) masked score (TDM-staged k tile; writes d_out region 1)
    score_kernel<<<B_ * 256 * 8, 256, 0, stream>>>(qB, kB, weights, out_masked);
    // 6) top-512 indices (writes d_out region 0)
    topk_kernel<<<B_ * S_, 512, 0, stream>>>(out_masked, out_idx);
}
